// CGCNNDynamics_17489106829435
// MI455X (gfx1250) — matrix operand order repaired and checked by reading, not compiled, not run on hardware
//
#include <hip/hip_runtime.h>

#define NN 2048
#define BB 4
#define KNN 4

typedef __attribute__((ext_vector_type(16))) __bf16 v16bf;
typedef __attribute__((ext_vector_type(8)))  float  v8f;

__device__ __forceinline__ unsigned short f2bf(float f) {
    unsigned u = __float_as_uint(f);
    u += 0x7FFFu + ((u >> 16) & 1u);          // round-to-nearest-even
    return (unsigned short)(u >> 16);
}

__device__ __forceinline__ float leakyf(float v) {
    return v >= 0.0f ? v : 0.2f * v;
}

// ---------------------------------------------------------------------------
// Generic fp32 -> bf16 conversion with optional column zero-padding.
// ---------------------------------------------------------------------------
__global__ void __launch_bounds__(256)
f32_to_bf16_pad(const float* __restrict__ src, unsigned short* __restrict__ dst,
                int rows, int srcC, int dstC)
{
    long total = (long)rows * dstC;
    for (long i = (long)blockIdx.x * blockDim.x + threadIdx.x; i < total;
         i += (long)gridDim.x * blockDim.x) {
        int r = (int)(i / dstC);
        int c = (int)(i % dstC);
        dst[i] = (c < srcC) ? f2bf(src[(long)r * srcC + c]) : (unsigned short)0;
    }
}

// ---------------------------------------------------------------------------
// Build X0 (B, 320, NN) bf16: rows 0..255 sampled^T, 256..263 action bcast,
// 264..319 zero pad (so GEMM K=320 is a multiple of 64).
// ---------------------------------------------------------------------------
__global__ void __launch_bounds__(256)
build_x0(const float* __restrict__ sampled, const float* __restrict__ action,
         unsigned short* __restrict__ X0)
{
    long total = (long)BB * 320 * NN;
    for (long i = (long)blockIdx.x * blockDim.x + threadIdx.x; i < total;
         i += (long)gridDim.x * blockDim.x) {
        int b   = (int)(i / (320L * NN));
        int rem = (int)(i - (long)b * 320 * NN);
        int r = rem / NN;
        int n = rem - r * NN;
        float v = 0.0f;
        if (r < 256)      v = sampled[((long)b * NN + n) * 256 + r];
        else if (r < 264) v = action[b * 8 + (r - 256)];
        X0[i] = f2bf(v);
    }
}

// ---------------------------------------------------------------------------
// Brute-force 4-NN over 2048 points per batch (squared distance, stable
// insertion so ties keep the lower index, matching jax top_k(-d)).
// ---------------------------------------------------------------------------
__global__ void __launch_bounds__(256)
knn_kernel(const float* __restrict__ center, int* __restrict__ idx)
{
    __shared__ float sx[256], sy[256], sz[256];
    const int b = blockIdx.y;
    const int q = blockIdx.x * 256 + threadIdx.x;
    const float* cb = center + (long)b * NN * 3;
    const float qx = cb[q * 3 + 0];
    const float qy = cb[q * 3 + 1];
    const float qz = cb[q * 3 + 2];

    float bd0 = 3.4e38f, bd1 = 3.4e38f, bd2 = 3.4e38f, bd3 = 3.4e38f;
    int   bi0 = 0, bi1 = 0, bi2 = 0, bi3 = 0;

    for (int t0 = 0; t0 < NN; t0 += 256) {
        int p = t0 + threadIdx.x;
        sx[threadIdx.x] = cb[p * 3 + 0];
        sy[threadIdx.x] = cb[p * 3 + 1];
        sz[threadIdx.x] = cb[p * 3 + 2];
        __syncthreads();
        for (int j = 0; j < 256; ++j) {
            float dx = qx - sx[j];
            float dy = qy - sy[j];
            float dz = qz - sz[j];
            float d = fmaf(dx, dx, fmaf(dy, dy, dz * dz));
            int pj = t0 + j;
            if (d < bd3) {
                if (d < bd2) {
                    bd3 = bd2; bi3 = bi2;
                    if (d < bd1) {
                        bd2 = bd1; bi2 = bi1;
                        if (d < bd0) { bd1 = bd0; bi1 = bi0; bd0 = d; bi0 = pj; }
                        else         { bd1 = d;   bi1 = pj; }
                    } else { bd2 = d; bi2 = pj; }
                } else { bd3 = d; bi3 = pj; }
            }
        }
        __syncthreads();
    }
    int* op = idx + ((long)b * NN + q) * KNN;
    op[0] = bi0; op[1] = bi1; op[2] = bi2; op[3] = bi3;
}

// ---------------------------------------------------------------------------
// Edge feature build: E (2*Cin x NN*4) bf16 = [ f[:,idx]-f[:,n] ; f[:,n] ]
// ---------------------------------------------------------------------------
__global__ void __launch_bounds__(256)
build_edge(const float* __restrict__ Fsrc, const int* __restrict__ idxq,
           unsigned short* __restrict__ E, int Cin)
{
    long total = (long)Cin * NN * KNN;
    for (long i = (long)blockIdx.x * blockDim.x + threadIdx.x; i < total;
         i += (long)gridDim.x * blockDim.x) {
        int c  = (int)(i / (NN * KNN));
        int nk = (int)(i % (NN * KNN));
        int n = nk >> 2;
        int j = idxq[nk];
        float fn = Fsrc[(long)c * NN + n];
        float fj = Fsrc[(long)c * NN + j];
        E[(long)c * NN * KNN + nk]          = f2bf(fj - fn);
        E[(long)(Cin + c) * NN * KNN + nk]  = f2bf(fn);
    }
}

// ---------------------------------------------------------------------------
// Double-buffered bf16 WMMA GEMM:
//   C(M x Ncols) fp32 = A(M x Kd) * B(Kd x Ncols) [+ bias]
// Block tile 128x128, stage BK=64 (2 WMMA K-chunks), 256 threads = 8 waves,
// each wave 32x64 (2x4 accumulator tiles), one s_barrier per 64-K stage.
// A tile: CDNA5 async copy global->LDS (global_load_async_to_lds_b128,
//   ASYNCcnt) into the alternate buffer -- no VGPR staging, no spills,
//   full overlap with the WMMA section; s_wait_asynccnt 0 before the barrier.
// B tile: VGPR path with v_perm_b32 transpose-pack into [n][k] layout.
// ---------------------------------------------------------------------------
template <bool HASBIAS>
__global__ void __launch_bounds__(256, 1)
gemm_bf16_wmma(const unsigned short* __restrict__ A,
               const unsigned short* __restrict__ Bm,
               float* __restrict__ C,
               const float* __restrict__ bias,
               int M, int Kd, int Ncols, long strideB, long strideC)
{
    __shared__ __align__(16) unsigned short As[2][128][72];
    __shared__ __align__(16) unsigned short Bs[2][128][72];

    const int tid  = threadIdx.x;
    const int lane = tid & 31;
    const int wave = tid >> 5;
    const int wm   = wave >> 1;    // 0..3 : 32-row slice of M
    const int wn   = wave & 1;     // 0..1 : 64-col slice of N
    const int l16  = lane & 15;
    const int lh   = lane >> 4;

    const int brow = blockIdx.y * 128;
    const int bcol = blockIdx.x * 128;
    const long zB = (long)blockIdx.z * strideB;
    const long zC = (long)blockIdx.z * strideC;

    const unsigned short* Ag = A + (long)brow * Kd;
    const unsigned short* Bg = Bm + zB + bcol;

    // staging maps:
    //   A tile 128x64: thread -> rows (tid>>3)+32r, cols (tid&7)*8 (async copy)
    //   B tile 64x128: thread -> k-pair 2*(tid>>4)+32r (k and k+1),
    //                  cols (tid&15)*8; v_perm-packed dword stores to Bs[n][k]
    const int arow = tid >> 3;          // 0..31 (+32*r)
    const int acol = (tid & 7) * 8;     // 0..56
    const int bk2  = (tid >> 4) * 2;    // 0..30 even (+32*r)
    const int bnc  = (tid & 15) * 8;    // 0..120

    // per-thread A offsets (elements) for the 4 async copies
    int aoff[4];
    #pragma unroll
    for (int r = 0; r < 4; ++r)
        aoff[r] = (arow + r * 32) * Kd + acol;

    v8f acc[2][4];
    #pragma unroll
    for (int i = 0; i < 2; ++i)
        #pragma unroll
        for (int j = 0; j < 4; ++j)
            #pragma unroll
            for (int e = 0; e < 8; ++e) acc[i][j][e] = 0.0f;

    uint4 gb[4];

    // async copy of the 128x64 A tile for K-offset k0 into As[dstbuf]
    auto asyncLoadA = [&](int k0, int dstbuf) {
        #pragma unroll
        for (int r = 0; r < 4; ++r) {
            unsigned lds  = (unsigned)(unsigned long)&As[dstbuf][arow + r * 32][acol];
            unsigned goff = (unsigned)((aoff[r] + k0) * 2);
            asm volatile("global_load_async_to_lds_b128 %0, %1, %2"
                         :: "v"(lds), "v"(goff), "s"(Ag) : "memory");
        }
    };
    auto waitAsync = [&]() {
        asm volatile("s_wait_asynccnt 0x0" ::: "memory");
    };

    auto loadGB = [&](int k0) {
        #pragma unroll
        for (int r = 0; r < 2; ++r) {
            gb[2 * r + 0] = *(const uint4*)(Bg + (long)(k0 + bk2 + 0 + r * 32) * Ncols + bnc);
            gb[2 * r + 1] = *(const uint4*)(Bg + (long)(k0 + bk2 + 1 + r * 32) * Ncols + bnc);
        }
    };
    auto storeSB = [&](int buf) {
        #pragma unroll
        for (int r = 0; r < 2; ++r) {
            const unsigned* lw = (const unsigned*)&gb[2 * r + 0]; // row k
            const unsigned* hw = (const unsigned*)&gb[2 * r + 1]; // row k+1
            int k = bk2 + r * 32;
            #pragma unroll
            for (int w = 0; w < 4; ++w) {
                // pack (k,k+1) bf16 pair per column with single v_perm_b32
                unsigned p0 = __builtin_amdgcn_perm(hw[w], lw[w], 0x05040100u);
                unsigned p1 = __builtin_amdgcn_perm(hw[w], lw[w], 0x07060302u);
                *(unsigned*)&Bs[buf][bnc + 2 * w + 0][k] = p0;
                *(unsigned*)&Bs[buf][bnc + 2 * w + 1][k] = p1;
            }
        }
    };

    auto mmaChunk = [&](int buf, int kk) {
        v16bf afrag[2], bfrag[4];
        #pragma unroll
        for (int i = 0; i < 2; ++i) {
            int row = wm * 32 + i * 16 + l16;
            union { uint4 q[2]; v16bf v; } u;
            u.q[0] = *(const uint4*)&As[buf][row][kk * 32 + lh * 8];
            u.q[1] = *(const uint4*)&As[buf][row][kk * 32 + 16 + lh * 8];
            afrag[i] = u.v;
        }
        #pragma unroll
        for (int j = 0; j < 4; ++j) {
            int col = wn * 64 + j * 16 + l16;
            union { uint4 q[2]; v16bf v; } u;
            u.q[0] = *(const uint4*)&Bs[buf][col][kk * 32 + lh * 8];
            u.q[1] = *(const uint4*)&Bs[buf][col][kk * 32 + 16 + lh * 8];
            bfrag[j] = u.v;
        }
        #pragma unroll
        for (int i = 0; i < 2; ++i)
            #pragma unroll
            for (int j = 0; j < 4; ++j)
                acc[i][j] = __builtin_amdgcn_wmma_f32_16x16x32_bf16(
                    false, afrag[i], false, bfrag[j],
                    (short)0, acc[i][j], false, false);
    };

    // prologue: stage 0 into buffer 0
    asyncLoadA(0, 0);
    loadGB(0);
    storeSB(0);
    waitAsync();
    __syncthreads();

    const int nStages = Kd >> 6;
    for (int s = 0; s < nStages; ++s) {
        const int buf = s & 1;
        const bool pre = (s + 1 < nStages);
        if (pre) {
            asyncLoadA((s + 1) << 6, buf ^ 1);   // background copy, no VGPRs
            loadGB((s + 1) << 6);
        }

        mmaChunk(buf, 0);
        mmaChunk(buf, 1);

        if (pre) {
            storeSB(buf ^ 1);                    // transpose-pack B prefetch
            waitAsync();                         // A tile landed in LDS
        }
        __syncthreads();
    }

    // preload bias (branch-free, compile-time)
    float bv[2][8];
    if (HASBIAS) {
        #pragma unroll
        for (int i = 0; i < 2; ++i)
            #pragma unroll
            for (int v = 0; v < 8; ++v)
                bv[i][v] = bias[brow + wm * 32 + i * 16 + lh * 8 + v];
    }

    // write back (C layout: lanes 0-15 -> M=v, lanes 16-31 -> M=8+v; N=lane&15)
    #pragma unroll
    for (int i = 0; i < 2; ++i) {
        #pragma unroll
        for (int j = 0; j < 4; ++j) {
            int col = bcol + wn * 64 + j * 16 + l16;
            #pragma unroll
            for (int v = 0; v < 8; ++v) {
                int row = brow + wm * 32 + i * 16 + lh * 8 + v;
                float val = acc[i][j][v];
                if (HASBIAS) val += bv[i][v];
                C[zC + (long)row * Ncols + col] = val;
            }
        }
    }
}

// ---------------------------------------------------------------------------
// Group-norm statistics: sum / sumsq per group (group rows are contiguous).
// ---------------------------------------------------------------------------
__global__ void __launch_bounds__(256)
zero_f32(float* __restrict__ p, int n)
{
    int i = blockIdx.x * blockDim.x + threadIdx.x;
    if (i < n) p[i] = 0.0f;
}

__global__ void __launch_bounds__(256)
gn_stats(const float* __restrict__ Y, float* __restrict__ stats,
         long groupElems, long groupStride)
{
    __shared__ float ssum[256], ssq[256];
    const long base = (long)blockIdx.y * groupStride;
    float s = 0.0f, ss = 0.0f;
    for (long i = (long)blockIdx.x * 256 + threadIdx.x; i < groupElems;
         i += (long)gridDim.x * 256) {
        float v = Y[base + i];
        s += v;
        ss = fmaf(v, v, ss);
    }
    ssum[threadIdx.x] = s;
    ssq[threadIdx.x]  = ss;
    __syncthreads();
    for (int o = 128; o > 0; o >>= 1) {
        if (threadIdx.x < (unsigned)o) {
            ssum[threadIdx.x] += ssum[threadIdx.x + o];
            ssq[threadIdx.x]  += ssq[threadIdx.x + o];
        }
        __syncthreads();
    }
    if (threadIdx.x == 0) {
        atomicAdd(&stats[blockIdx.y * 2 + 0], ssum[0]);
        atomicAdd(&stats[blockIdx.y * 2 + 1], ssq[0]);
    }
}

// ---------------------------------------------------------------------------
// Edge block epilogue: normalize, affine, leaky-relu, max over K=4.
// ---------------------------------------------------------------------------
__global__ void __launch_bounds__(256)
gn_apply_max(const float* __restrict__ Y, const float* __restrict__ stats,
             const float* __restrict__ gw, const float* __restrict__ gb,
             float* __restrict__ Fout, int O, int cg)
{
    long total = (long)O * NN;
    float cnt = (float)cg * NN * KNN;
    for (long i = (long)blockIdx.x * blockDim.x + threadIdx.x; i < total;
         i += (long)gridDim.x * blockDim.x) {
        int o = (int)(i / NN);
        int n = (int)(i % NN);
        int g = o / cg;
        float mean = stats[g * 2 + 0] / cnt;
        float var  = stats[g * 2 + 1] / cnt - mean * mean;
        float rstd = rsqrtf(var + 1e-5f);
        float sc = gw[o] * rstd;
        float sh = gb[o] - mean * sc;
        const float* yp = Y + ((long)o * NN + n) * KNN;
        float m = -3.4e38f;
        #pragma unroll
        for (int k = 0; k < KNN; ++k) {
            float v = leakyf(fmaf(yp[k], sc, sh));
            m = fmaxf(m, v);
        }
        Fout[(long)o * NN + n] = m;
    }
}

// ---------------------------------------------------------------------------
// Final epilogue: normalize + affine + leaky, write transposed (n, o).
// ---------------------------------------------------------------------------
__global__ void __launch_bounds__(256)
final_apply(const float* __restrict__ Y5b, const float* __restrict__ stats,
            const float* __restrict__ gw, const float* __restrict__ gb,
            float* __restrict__ outb)
{
    const int O = 512, cg = 128;
    long total = (long)O * NN;
    float cnt = (float)cg * NN;
    for (long i = (long)blockIdx.x * blockDim.x + threadIdx.x; i < total;
         i += (long)gridDim.x * blockDim.x) {
        int o = (int)(i / NN);
        int n = (int)(i % NN);
        int g = o / cg;
        float mean = stats[g * 2 + 0] / cnt;
        float var  = stats[g * 2 + 1] / cnt - mean * mean;
        float rstd = rsqrtf(var + 1e-5f);
        float sc = gw[o] * rstd;
        float v = fmaf(Y5b[(long)o * NN + n], sc, gb[o] - mean * sc);
        outb[(long)n * O + o] = leakyf(v);
    }
}

// ---------------------------------------------------------------------------
static inline int gsz(long total)
{
    long b = (total + 255) / 256;
    if (b > 4096) b = 4096;
    if (b < 1) b = 1;
    return (int)b;
}

extern "C" void kernel_launch(void* const* d_in, const int* in_sizes, int n_in,
                              void* d_out, int out_size, void* d_ws, size_t ws_size,
                              hipStream_t stream)
{
    (void)in_sizes; (void)n_in; (void)out_size; (void)ws_size;
    const float* sampled = (const float*)d_in[0];
    const float* center  = (const float*)d_in[1];
    const float* action  = (const float*)d_in[2];
    const float* w_in = (const float*)d_in[3];
    const float* b_in = (const float*)d_in[4];
    const float* w1  = (const float*)d_in[5];
    const float* g1w = (const float*)d_in[6];
    const float* g1b = (const float*)d_in[7];
    const float* w2  = (const float*)d_in[8];
    const float* g2w = (const float*)d_in[9];
    const float* g2b = (const float*)d_in[10];
    const float* w3  = (const float*)d_in[11];
    const float* g3w = (const float*)d_in[12];
    const float* g3b = (const float*)d_in[13];
    const float* w4  = (const float*)d_in[14];
    const float* g4w = (const float*)d_in[15];
    const float* g4b = (const float*)d_in[16];
    const float* w5  = (const float*)d_in[17];
    const float* b5  = (const float*)d_in[18];
    const float* g5w = (const float*)d_in[19];
    const float* g5b = (const float*)d_in[20];
    float* out = (float*)d_out;

    char* ws = (char*)d_ws;
    size_t off = 0;
    auto take = [&](size_t bytes) {
        size_t o = off;
        off += (bytes + 255) & ~(size_t)255;
        return o;
    };

    unsigned short* wbf_in = (unsigned short*)(ws + take((size_t)128 * 320 * 2));
    unsigned short* wbf1   = (unsigned short*)(ws + take((size_t)256 * 256 * 2));
    unsigned short* wbf2   = (unsigned short*)(ws + take((size_t)512 * 512 * 2));
    unsigned short* wbf3   = (unsigned short*)(ws + take((size_t)512 * 1024 * 2));
    unsigned short* wbf4   = (unsigned short*)(ws + take((size_t)1024 * 1024 * 2));
    unsigned short* wbf5   = (unsigned short*)(ws + take((size_t)512 * 2304 * 2));
    unsigned short* X0bf   = (unsigned short*)(ws + take((size_t)BB * 320 * NN * 2));
    float* F0   = (float*)(ws + take((size_t)BB * 128 * NN * 4));
    int*   idxb = (int*)  (ws + take((size_t)BB * NN * KNN * 4));
    float* F    = (float*)(ws + take((size_t)BB * 2304 * NN * 4));
    unsigned short* FCbf = (unsigned short*)(ws + take((size_t)BB * 2304 * NN * 2));
    unsigned short* Ebuf = (unsigned short*)(ws + take((size_t)1024 * NN * KNN * 2));
    float* Ybuf  = (float*)(ws + take((size_t)1024 * NN * KNN * 4));
    float* Y5    = (float*)(ws + take((size_t)BB * 512 * NN * 4));
    float* stats = (float*)(ws + take(256));

    // ---- weight conversion (fp32 -> bf16, w_in K-padded 264 -> 320) ----
    f32_to_bf16_pad<<<gsz(128L * 320), 256, 0, stream>>>(w_in, wbf_in, 128, 264, 320);
    f32_to_bf16_pad<<<gsz(256L * 256), 256, 0, stream>>>(w1, wbf1, 256, 256, 256);
    f32_to_bf16_pad<<<gsz(512L * 512), 256, 0, stream>>>(w2, wbf2, 512, 512, 512);
    f32_to_bf16_pad<<<gsz(512L * 1024), 256, 0, stream>>>(w3, wbf3, 512, 1024, 1024);
    f32_to_bf16_pad<<<gsz(1024L * 1024), 256, 0, stream>>>(w4, wbf4, 1024, 1024, 1024);
    f32_to_bf16_pad<<<gsz(512L * 2304), 256, 0, stream>>>(w5, wbf5, 512, 2304, 2304);

    // ---- input features + kNN ----
    build_x0<<<gsz((long)BB * 320 * NN), 256, 0, stream>>>(sampled, action, X0bf);
    knn_kernel<<<dim3(NN / 256, BB), 256, 0, stream>>>(center, idxb);

    // f0 = w_in @ x + b_in   (M=128, K=320, N=2048, per batch via grid.z)
    {
        dim3 g(NN / 128, 1, BB);
        gemm_bf16_wmma<true><<<g, 256, 0, stream>>>(wbf_in, X0bf, F0, b_in,
                                                    128, 320, NN,
                                                    (long)320 * NN, (long)128 * NN);
    }

    // ---- four edge blocks ----
    const int  cins[4]   = {128, 256, 512, 512};
    const int  outsO[4]  = {256, 512, 512, 1024};
    const long inRow[4]  = {0, 0, 256, 768};
    const long outRow[4] = {0, 256, 768, 1280};
    const unsigned short* Wb[4] = {wbf1, wbf2, wbf3, wbf4};
    const float* gws[4] = {g1w, g2w, g3w, g4w};
    const float* gbs[4] = {g1b, g2b, g3b, g4b};

    for (int blk = 0; blk < 4; ++blk) {
        const int Cin = cins[blk];
        const int O   = outsO[blk];
        const int cg  = O / 4;
        for (int b = 0; b < BB; ++b) {
            const float* Fsrc = (blk == 0)
                ? (F0 + (long)b * 128 * NN)
                : (F + (long)b * 2304 * NN + inRow[blk] * NN);
            const int* iq = idxb + (long)b * NN * KNN;

            build_edge<<<gsz((long)Cin * NN * KNN), 256, 0, stream>>>(Fsrc, iq, Ebuf, Cin);

            dim3 g(NN * KNN / 128, O / 128, 1);
            gemm_bf16_wmma<false><<<g, 256, 0, stream>>>(Wb[blk], Ebuf, Ybuf, nullptr,
                                                         O, 2 * Cin, NN * KNN, 0, 0);

            zero_f32<<<1, 32, 0, stream>>>(stats, 8);
            gn_stats<<<dim3(64, 4), 256, 0, stream>>>(Ybuf, stats,
                                                      (long)cg * NN * KNN,
                                                      (long)cg * NN * KNN);
            float* Fdst = F + (long)b * 2304 * NN + outRow[blk] * NN;
            gn_apply_max<<<gsz((long)O * NN), 256, 0, stream>>>(Ybuf, stats,
                                                                gws[blk], gbs[blk],
                                                                Fdst, O, cg);
        }
    }

    // ---- final conv: out = leaky(gn(w5 @ fc + b5)) ----
    f32_to_bf16_pad<<<gsz((long)BB * 2304 * NN), 256, 0, stream>>>(F, FCbf,
                                                                   BB * 2304, NN, NN);
    {
        dim3 g(NN / 128, 512 / 128, BB);
        gemm_bf16_wmma<true><<<g, 256, 0, stream>>>(wbf5, FCbf, Y5, b5,
                                                    512, 2304, NN,
                                                    (long)2304 * NN, (long)512 * NN);
    }
    for (int b = 0; b < BB; ++b) {
        zero_f32<<<1, 32, 0, stream>>>(stats, 8);
        gn_stats<<<dim3(32, 4), 256, 0, stream>>>(Y5 + (long)b * 512 * NN, stats,
                                                  (long)128 * NN, (long)128 * NN);
        final_apply<<<gsz(512L * NN), 256, 0, stream>>>(Y5 + (long)b * 512 * NN, stats,
                                                        g5w, g5b,
                                                        out + (long)b * NN * 512);
    }
}
